// BiAttention_73383811219841
// MI455X (gfx1250) — matrix operand order repaired and checked by reading, not compiled, not run on hardware
//
#include <hip/hip_runtime.h>
#include <hip/hip_bf16.h>

typedef __bf16 bf16_t;
typedef __attribute__((ext_vector_type(16))) __bf16 v16bf;
typedef __attribute__((ext_vector_type(8)))  __bf16 v8bf;
typedef __attribute__((ext_vector_type(4)))  __bf16 v4bf;
typedef __attribute__((ext_vector_type(8)))  float  v8f;
typedef __attribute__((ext_vector_type(4)))  float  v4f;

#define NEG_INF (-1e30f)

__device__ inline v8f vzero8() {
    v8f z;
    for (int i = 0; i < 8; ++i) z[i] = 0.0f;
    return z;
}

// --- CDNA5 async global->LDS copy (16 bytes per lane), tracked by ASYNCcnt ---
__device__ inline void async_copy_b128(unsigned lds_off, const void* gptr) {
    asm volatile("global_load_async_to_lds_b128 %0, %1, off"
                 :: "v"(lds_off), "v"(gptr)
                 : "memory");
}
__device__ inline void wait_async_le5() {
    asm volatile("s_wait_asynccnt 5" ::: "memory");
}
__device__ inline void wait_async_0() {
    asm volatile("s_wait_asynccnt 0" ::: "memory");
}
__device__ inline unsigned lds_off(const void* p) {
    return (unsigned)(unsigned long long)p;   // addrspace(3) ptr -> byte offset
}

// A fragment: 16x32 bf16, M x K. lanes 0-15 hold K=0..7,16..23 ; lanes 16-31 K=8..15,24..31
__device__ inline v16bf load_a_frag(const bf16_t* lds, int lda, int m0, int k0, int lane) {
    int m    = m0 + (lane & 15);
    int half = lane >> 4;
    const bf16_t* p = lds + (size_t)m * lda + k0;
    v8bf c0 = *(const v8bf*)(p + half * 8);
    v8bf c1 = *(const v8bf*)(p + 16 + half * 8);
    return __builtin_shufflevector(c0, c1, 0,1,2,3,4,5,6,7,8,9,10,11,12,13,14,15);
}

// B fragment: 32x16 bf16 (K x N), weights stored row-major as N rows x K cols.
__device__ inline v16bf load_b_frag(const bf16_t* lds, int ldb, int n0, int k0, int lane) {
    int n    = n0 + (lane & 15);
    int half = lane >> 4;
    const bf16_t* p = lds + (size_t)n * ldb + k0;
    v8bf c0 = *(const v8bf*)(p + half * 8);
    v8bf c1 = *(const v8bf*)(p + 16 + half * 8);
    return __builtin_shufflevector(c0, c1, 0,1,2,3,4,5,6,7,8,9,10,11,12,13,14,15);
}

__device__ inline v8f wmma_bf16(v16bf a, v16bf b, v8f c) {
    return __builtin_amdgcn_wmma_f32_16x16x32_bf16(false, a, false, b, (short)0, c, false, false);
}

__device__ inline v4bf cvt4(v4f f) {
    v4bf o;
    o[0] = (bf16_t)f[0]; o[1] = (bf16_t)f[1];
    o[2] = (bf16_t)f[2]; o[3] = (bf16_t)f[3];
    return o;
}

// ---------------------------------------------------------------- K0: f32 -> bf16 (4/thread)
__global__ void cvt_bf16_kernel(const float* __restrict__ src, bf16_t* __restrict__ dst, int n) {
    int i = (blockIdx.x * 256 + threadIdx.x) * 4;
    if (i < n) *(v4bf*)&dst[i] = cvt4(*(const v4f*)&src[i]);
}

// ---------------------------------------------------------------- K1: Y = leaky_relu(X @ W1^T + b1), bf16 out
// Tile 64 x 128, 8 waves (2 x 4), wave 32x32 (2x2 wmma tiles), K = 512
__global__ void gemm_w1_act_kernel(const float* __restrict__ X, const bf16_t* __restrict__ Wb,
                                   const float* __restrict__ bias, bf16_t* __restrict__ Y) {
    __shared__ bf16_t sA[64][32];
    __shared__ bf16_t sB[128][32];
    int tid = threadIdx.x, lane = tid & 31, wave = tid >> 5;
    int wm = wave >> 2, wn = wave & 3;
    size_t row0 = (size_t)blockIdx.x * 64;
    int    col0 = blockIdx.y * 128;

    v8f acc[4];
    for (int i = 0; i < 4; ++i) acc[i] = vzero8();

    for (int k0 = 0; k0 < 512; k0 += 32) {
        // B tile: async global->LDS, 128 rows x 4 chunks of 16B = 512 chunks (2/thread)
        for (int i = tid; i < 512; i += 256) {
            int r = i >> 2, c = (i & 3) * 8;
            async_copy_b128(lds_off(&sB[r][c]), &Wb[(size_t)(col0 + r) * 512 + k0 + c]);
        }
        // A tile: f32 -> bf16 through VGPRs, vectorized (float4 -> v4bf, 2/thread)
        for (int i = tid; i < 512; i += 256) { // 64 rows x 8 quads
            int r = i >> 3, c = (i & 7) * 4;
            *(v4bf*)&sA[r][c] = cvt4(*(const v4f*)&X[(row0 + r) * 512 + k0 + c]);
        }
        wait_async_0();
        __syncthreads();
        for (int mi = 0; mi < 2; ++mi) {
            v16bf a = load_a_frag(&sA[0][0], 32, wm * 32 + mi * 16, 0, lane);
            for (int ni = 0; ni < 2; ++ni) {
                v16bf b = load_b_frag(&sB[0][0], 32, wn * 32 + ni * 16, 0, lane);
                acc[mi * 2 + ni] = wmma_bf16(a, b, acc[mi * 2 + ni]);
            }
        }
        __syncthreads();
    }
    for (int mi = 0; mi < 2; ++mi)
        for (int ni = 0; ni < 2; ++ni) {
            int n = col0 + wn * 32 + ni * 16 + (lane & 15);
            float bv = bias[n];
            for (int v = 0; v < 8; ++v) {
                size_t m = row0 + wm * 32 + mi * 16 + (lane >> 4) * 8 + v;
                float val = acc[mi * 2 + ni][v] + bv;
                val = val > 0.0f ? val : 0.01f * val;
                Y[m * 512 + n] = (bf16_t)val;
            }
        }
}

// ---------------------------------------------------------------- K2: fused masked-softmax attention
// Per block: 32 C-rows. S = C_ @ Q_^T (32x512), mask, softmax, P @ Qraw -> attn (32x512).
// Dynamic LDS map (bytes):
//   [0,2048)        sA   : 32x32 bf16
//   [2048,67584)    union: sQ 512x32 bf16 | sS 32x512 f32 | sBt 512x32 bf16
//   [67584,100352)  sP   : 32x512 bf16
//   [100352,102400) red  : 512 f32
__global__ void attention_fused_kernel(const bf16_t* __restrict__ Cb, const bf16_t* __restrict__ Qb,
                                       const bf16_t* __restrict__ Qraw,
                                       const float* __restrict__ Cf,
                                       const int* __restrict__ Cmask, const int* __restrict__ Qmask,
                                       float* __restrict__ attn, bf16_t* __restrict__ cat) {
    extern __shared__ char smem[];
    bf16_t* sA  = (bf16_t*)smem;
    bf16_t* sQ  = (bf16_t*)(smem + 2048);
    float*  sS  = (float*)(smem + 2048);
    bf16_t* sBt = (bf16_t*)(smem + 2048);
    bf16_t* sP  = (bf16_t*)(smem + 67584);
    float*  red = (float*)(smem + 100352);

    int tid = threadIdx.x, lane = tid & 31, wave = tid >> 5;
    int b   = blockIdx.x >> 6;
    int rt  = blockIdx.x & 63;
    int row0 = rt * 32;
    size_t grow0 = (size_t)b * 2048 + row0;

    // ---- Phase 1: S = C_ @ Q_^T (async-staged tiles)
    v8f acc[8];
    for (int i = 0; i < 8; ++i) acc[i] = vzero8();
    for (int k0 = 0; k0 < 512; k0 += 32) {
        if (tid < 128) { // 32 rows x 4 chunks
            int r = tid >> 2, c = (tid & 3) * 8;
            async_copy_b128(lds_off(&sA[r * 32 + c]), &Cb[(grow0 + r) * 512 + k0 + c]);
        }
        for (int i = tid; i < 2048; i += 256) { // 512 q-rows x 4 chunks
            int r = i >> 2, c = (i & 3) * 8;
            async_copy_b128(lds_off(&sQ[r * 32 + c]), &Qb[((size_t)b * 512 + r) * 512 + k0 + c]);
        }
        wait_async_0();
        __syncthreads();
        for (int mi = 0; mi < 2; ++mi) {
            v16bf a = load_a_frag(sA, 32, mi * 16, 0, lane);
            for (int ni = 0; ni < 4; ++ni) {
                v16bf bb = load_b_frag(sQ, 32, wave * 64 + ni * 16, 0, lane);
                acc[mi * 4 + ni] = wmma_bf16(a, bb, acc[mi * 4 + ni]);
            }
        }
        __syncthreads();
    }

    // ---- Phase 2: masks, spill S to LDS (f32)
    for (int mi = 0; mi < 2; ++mi)
        for (int ni = 0; ni < 4; ++ni) {
            int n = wave * 64 + ni * 16 + (lane & 15);
            int qm = Qmask[b * 512 + n];
            for (int v = 0; v < 8; ++v) {
                int m = mi * 16 + (lane >> 4) * 8 + v;
                int cm = Cmask[grow0 + m];
                float s = acc[mi * 4 + ni][v];
                sS[m * 512 + n] = (cm && qm) ? s : NEG_INF;
            }
        }
    __syncthreads();

    // ---- Phase 3: softmax over 512 cols (8 threads/row, 64 cols each)
    {
        int r = tid >> 3, sg = tid & 7, c0 = sg * 64;
        float mx = -3.0e38f;
        for (int c = 0; c < 64; ++c) mx = fmaxf(mx, sS[r * 512 + c0 + c]);
        red[r * 8 + sg] = mx;
        __syncthreads();
        float rmx = red[r * 8];
        for (int j = 1; j < 8; ++j) rmx = fmaxf(rmx, red[r * 8 + j]);
        float sum = 0.0f;
        for (int c = 0; c < 64; ++c) {
            float e = __expf(sS[r * 512 + c0 + c] - rmx);
            sS[r * 512 + c0 + c] = e;
            sum += e;
        }
        red[256 + r * 8 + sg] = sum;
        __syncthreads();
        float rs = 0.0f;
        for (int j = 0; j < 8; ++j) rs += red[256 + r * 8 + j];
        float inv = 1.0f / rs;
        for (int c = 0; c < 64; ++c)
            sP[r * 512 + c0 + c] = (bf16_t)(sS[r * 512 + c0 + c] * inv);
    }
    __syncthreads();

    // ---- Phase 4: attn = P @ Qraw (Qraw transposed through LDS)
    v8f acc2[8];
    for (int i = 0; i < 8; ++i) acc2[i] = vzero8();
    for (int q0 = 0; q0 < 512; q0 += 32) {
        for (int i = tid; i < 2048; i += 256) { // 32 q-rows x 64 chunks along d
            int qq = i >> 6, dc = (i & 63) * 8;
            v8bf ch = *(const v8bf*)&Qraw[((size_t)b * 512 + q0 + qq) * 512 + dc];
            for (int e = 0; e < 8; ++e) sBt[(dc + e) * 32 + qq] = ch[e];
        }
        __syncthreads();
        for (int mi = 0; mi < 2; ++mi) {
            v16bf a = load_a_frag(sP, 512, mi * 16, q0, lane);
            for (int ni = 0; ni < 4; ++ni) {
                v16bf bb = load_b_frag(sBt, 32, wave * 64 + ni * 16, 0, lane);
                acc2[mi * 4 + ni] = wmma_bf16(a, bb, acc2[mi * 4 + ni]);
            }
        }
        __syncthreads();
    }

    // ---- Phase 5: emit attn_C (f32) and cat segments (bf16)
    for (int mi = 0; mi < 2; ++mi)
        for (int ni = 0; ni < 4; ++ni) {
            int d = wave * 64 + ni * 16 + (lane & 15);
            for (int v = 0; v < 8; ++v) {
                int m = mi * 16 + (lane >> 4) * 8 + v;
                size_t gr = grow0 + m;
                float a = acc2[mi * 4 + ni][v];
                float c = Cf[gr * 512 + d];
                attn[gr * 512 + d] = a;
                size_t cr = gr * 2048;
                cat[cr + d]        = (bf16_t)c;
                cat[cr + 512 + d]  = (bf16_t)a;
                cat[cr + 1024 + d] = (bf16_t)(a - c);
                cat[cr + 1536 + d] = (bf16_t)(a * c);
            }
        }
}

// ---------------------------------------------------------------- K3: dual GEMM + gated epilogue
// Double-buffered async-to-LDS staging: 5 async b128 per thread per stage.
__device__ inline void k3_prefetch(const bf16_t* __restrict__ cat,
                                   const bf16_t* __restrict__ Wfb, const bf16_t* __restrict__ Wgb,
                                   size_t row0, int col0, int k0, int tid,
                                   bf16_t (*sA)[32], bf16_t (*sF)[32], bf16_t (*sG)[32]) {
    { // A: 64 rows x 4 chunks -> exactly 256 chunks (1/thread)
        int r = tid >> 2, c = (tid & 3) * 8;
        async_copy_b128(lds_off(&sA[r][c]), &cat[(row0 + r) * 2048 + k0 + c]);
    }
    for (int i = tid; i < 512; i += 256) { // 128 rows x 4 chunks, both weight mats (2+2/thread)
        int r = i >> 2, c = (i & 3) * 8;
        async_copy_b128(lds_off(&sF[r][c]), &Wfb[(size_t)(col0 + r) * 2048 + k0 + c]);
        async_copy_b128(lds_off(&sG[r][c]), &Wgb[(size_t)(col0 + r) * 2048 + k0 + c]);
    }
}

__global__ void fuse_gate_gemm_kernel(const bf16_t* __restrict__ cat,
                                      const bf16_t* __restrict__ Wfb, const bf16_t* __restrict__ Wgb,
                                      const float* __restrict__ bfp, const float* __restrict__ bgp,
                                      const float* __restrict__ Cf, const float* __restrict__ attn,
                                      const int* __restrict__ Cmask, float* __restrict__ out) {
    __shared__ bf16_t sA[2][64][32];
    __shared__ bf16_t sF[2][128][32];
    __shared__ bf16_t sG[2][128][32];
    int tid = threadIdx.x, lane = tid & 31, wave = tid >> 5;
    int wm = wave >> 2, wn = wave & 3;
    size_t row0 = (size_t)blockIdx.x * 64;
    int    col0 = blockIdx.y * 128;

    v8f accF[4], accG[4];
    for (int i = 0; i < 4; ++i) { accF[i] = vzero8(); accG[i] = vzero8(); }

    k3_prefetch(cat, Wfb, Wgb, row0, col0, 0, tid, sA[0], sF[0], sG[0]);

    int buf = 0;
    for (int k0 = 0; k0 < 2048; k0 += 32, buf ^= 1) {
        bool more = (k0 + 32) < 2048;
        if (more)
            k3_prefetch(cat, Wfb, Wgb, row0, col0, k0 + 32, tid, sA[buf ^ 1], sF[buf ^ 1], sG[buf ^ 1]);
        if (more) wait_async_le5();   // current stage's 5 copies done; next stage in flight
        else      wait_async_0();
        __syncthreads();
        for (int mi = 0; mi < 2; ++mi) {
            v16bf a = load_a_frag(&sA[buf][0][0], 32, wm * 32 + mi * 16, 0, lane);
            for (int ni = 0; ni < 2; ++ni) {
                v16bf bF = load_b_frag(&sF[buf][0][0], 32, wn * 32 + ni * 16, 0, lane);
                accF[mi * 2 + ni] = wmma_bf16(a, bF, accF[mi * 2 + ni]);
                v16bf bG = load_b_frag(&sG[buf][0][0], 32, wn * 32 + ni * 16, 0, lane);
                accG[mi * 2 + ni] = wmma_bf16(a, bG, accG[mi * 2 + ni]);
            }
        }
        __syncthreads();
    }

    for (int mi = 0; mi < 2; ++mi)
        for (int ni = 0; ni < 2; ++ni) {
            int col = col0 + wn * 32 + ni * 16 + (lane & 15);
            int seg = col >> 9, d = col & 511;
            float bfv = bfp[col], bgv = bgp[col];
            for (int v = 0; v < 8; ++v) {
                size_t gr = row0 + wm * 32 + mi * 16 + (lane >> 4) * 8 + v;
                float f = tanhf(accF[mi * 2 + ni][v] + bfv);
                float g = 1.0f / (1.0f + __expf(-(accG[mi * 2 + ni][v] + bgv)));
                float c = Cf[gr * 512 + d];
                float a = attn[gr * 512 + d];
                float catv = (seg == 0) ? c : (seg == 1) ? a : (seg == 2) ? (a - c) : (a * c);
                float o = g * f + (1.0f - g) * catv;
                int cm = Cmask[gr];
                out[gr * 2048 + col] = cm ? o : NEG_INF;
            }
        }
}

// ---------------------------------------------------------------- host
extern "C" void kernel_launch(void* const* d_in, const int* in_sizes, int n_in,
                              void* d_out, int out_size, void* d_ws, size_t ws_size,
                              hipStream_t stream) {
    const float* C     = (const float*)d_in[0];
    const float* Q     = (const float*)d_in[1];
    const int*   Cmask = (const int*)d_in[2];
    const int*   Qmask = (const int*)d_in[3];
    const float* W1    = (const float*)d_in[4];
    const float* b1    = (const float*)d_in[5];
    const float* Wf    = (const float*)d_in[6];
    const float* bfp   = (const float*)d_in[7];
    const float* Wg    = (const float*)d_in[8];
    const float* bgp   = (const float*)d_in[9];
    float* out = (float*)d_out;

    char* ws = (char*)d_ws;
    bf16_t* W1b  = (bf16_t*)(ws + 0);                     //   512*512*2 =    524288
    bf16_t* Wfb  = (bf16_t*)(ws + 524288);                //  2048*2048*2 =  8388608
    bf16_t* Wgb  = (bf16_t*)(ws + 8912896);               //               8388608
    bf16_t* Qraw = (bf16_t*)(ws + 17301504);              // 16*512*512*2 = 8388608
    bf16_t* Cb   = (bf16_t*)(ws + 25690112);              // 32768*512*2 = 33554432
    bf16_t* Qb   = (bf16_t*)(ws + 59244544);              //  8192*512*2 =  8388608
    float*  attn = (float*)(ws + 67633152);               // 32768*512*4 = 67108864
    bf16_t* cat  = (bf16_t*)(ws + 134742016);             // 32768*2048*2 = 134217728

    // K0: convert weights and raw Q to bf16 (4 elems/thread)
    cvt_bf16_kernel<<<dim3(262144 / 1024), dim3(256), 0, stream>>>(W1, W1b, 262144);
    cvt_bf16_kernel<<<dim3(4194304 / 1024), dim3(256), 0, stream>>>(Wf, Wfb, 4194304);
    cvt_bf16_kernel<<<dim3(4194304 / 1024), dim3(256), 0, stream>>>(Wg, Wgb, 4194304);
    cvt_bf16_kernel<<<dim3(4194304 / 1024), dim3(256), 0, stream>>>(Q, Qraw, 4194304);

    // K1: C_ = leaky(C@W1^T+b1), Q_ = leaky(Q@W1^T+b1), bf16
    gemm_w1_act_kernel<<<dim3(32768 / 64, 4), dim3(256), 0, stream>>>(C, W1b, b1, Cb);
    gemm_w1_act_kernel<<<dim3(8192 / 64, 4), dim3(256), 0, stream>>>(Q, W1b, b1, Qb);

    // K2: fused masked-softmax attention -> attn (f32) + cat (bf16)
    attention_fused_kernel<<<dim3(16 * 64), dim3(256), 102400, stream>>>(
        Cb, Qb, Qraw, C, Cmask, Qmask, attn, cat);

    // K3: dual GEMM (Wf, Wg) + tanh/sigmoid gating + mask -> out
    fuse_gate_gemm_kernel<<<dim3(32768 / 64, 2048 / 128), dim3(256), 0, stream>>>(
        cat, Wfb, Wgb, bfp, bgp, C, attn, Cmask, out);

    (void)in_sizes; (void)n_in; (void)out_size; (void)ws_size;
}
